// CausalSelfAttention_68006512164945
// MI455X (gfx1250) — matrix operand order repaired and checked by reading, not compiled, not run on hardware
//
#include <hip/hip_runtime.h>
#include <hip/hip_bf16.h>

// ---- WMMA types (gfx1250, wave32) ----
typedef __attribute__((ext_vector_type(16))) _Float16 v16h;
typedef __attribute__((ext_vector_type(8)))  _Float16 v8h;
typedef __attribute__((ext_vector_type(4)))  _Float16 v4h;
typedef __attribute__((ext_vector_type(8)))  float    v8f;

#define N_B 4
#define N_T 2048
#define N_C 512
#define N_H 8
#define N_HD 64
#define N_3C 1536
#define NEG_INF_F (-1.0e30f)

__device__ __forceinline__ v8f wmma_f16(v16h a, v16h b, v8f c) {
  // D = A(16x32 f16) x B(32x16 f16) + C(16x16 f32)
  return __builtin_amdgcn_wmma_f32_16x16x32_f16(
      false, a, false, b, (short)0, c, false, false);
}

// two 8-half (16B) chunks -> v16h
__device__ __forceinline__ v16h load16h(const _Float16* __restrict__ pa,
                                        const _Float16* __restrict__ pb) {
  union { v16h v; v8h h[2]; } u;
  u.h[0] = *(const v8h*)pa;
  u.h[1] = *(const v8h*)pb;
  return u.v;
}

// ============================================================================
// Kernel 0: one-shot f32 -> f16 conversion (x, W_qkv, W_out), 4 elems/thread.
// Removes all per-tile v_cvt work from the GEMM inner loops.
// ============================================================================
__global__ __launch_bounds__(256) void cvt_f16_kernel(
    const float* __restrict__ src, _Float16* __restrict__ dst, int n4) {
  const int i = blockIdx.x * 256 + threadIdx.x;
  if (i < n4) {
    float4 f = ((const float4*)src)[i];
    v4h o;
    o[0] = (_Float16)f.x; o[1] = (_Float16)f.y;
    o[2] = (_Float16)f.z; o[3] = (_Float16)f.w;
    ((v4h*)dst)[i] = o;
  }
}

// ============================================================================
// Kernel 1: qkv = x @ W_qkv^T   (M=8192, N=1536, K=512), all-f16 operands
// Block = 256 threads (8 waves); wave computes 32 rows x 64 cols
// (2 A-frags reuse 4 B-frags -> 8 independent WMMAs per k-step).
// grid = (M/32, N/512)
// ============================================================================
__global__ __launch_bounds__(256) void qkv_gemm_kernel(
    const _Float16* __restrict__ X, const _Float16* __restrict__ W,
    _Float16* __restrict__ Y) {
  const int lane = threadIdx.x & 31;
  const int wave = threadIdx.x >> 5;
  const int grp  = lane >> 4;   // 0 or 1
  const int l16  = lane & 15;
  const int row0 = blockIdx.x * 32;
  const int col0 = blockIdx.y * 512 + wave * 64;

  v8f acc0[4] = {}, acc1[4] = {};
  const _Float16* xrow0 = X + (size_t)(row0 + l16) * N_C;
  const _Float16* xrow1 = X + (size_t)(row0 + 16 + l16) * N_C;

  for (int k0 = 0; k0 < N_C; k0 += 32) {
    __builtin_prefetch(xrow0 + k0 + 128, 0, 1);   // global_prefetch_b8
    // A fragments: 16x32 f16 layout (lane<16: K 0-7,16-23; lane>=16: 8-15,24-31)
    v16h a0 = load16h(xrow0 + k0 + grp * 8, xrow0 + k0 + 16 + grp * 8);
    v16h a1 = load16h(xrow1 + k0 + grp * 8, xrow1 + k0 + 16 + grp * 8);
#pragma unroll
    for (int n = 0; n < 4; ++n) {
      // B fragment: lane -> output column m (row m of W); grp selects K 8-blocks (same as A)
      const _Float16* wrow =
          W + (size_t)(col0 + n * 16 + l16) * N_C + k0;
      v16h b = load16h(wrow + grp * 8, wrow + 16 + grp * 8);
      acc0[n] = wmma_f16(a0, b, acc0[n]);
      acc1[n] = wmma_f16(a1, b, acc1[n]);
    }
  }
  // C/D layout: VGPR v, lane group -> row v + 8*grp, col l16
#pragma unroll
  for (int n = 0; n < 4; ++n) {
#pragma unroll
    for (int v = 0; v < 8; ++v) {
      const int r = row0 + v + grp * 8;
      const int c = col0 + n * 16 + l16;
      Y[(size_t)r * N_3C + c] = (_Float16)acc0[n][v];
      Y[(size_t)(r + 16) * N_3C + c] = (_Float16)acc1[n][v];
    }
  }
}

// ============================================================================
// Kernel 2: RoPE on q,k + repack:
//   Qh,Kh: [B,H,T,hd] f16 ;  Vt: [B,H,hd,T] f16 (transposed for PV B-matrix)
// One thread per (b,h,t,d). grid*block = B*H*T*hd = 4194304
// ============================================================================
__global__ __launch_bounds__(256) void rope_repack_kernel(
    const _Float16* __restrict__ qkv, const float* __restrict__ cosT,
    const float* __restrict__ sinT, _Float16* __restrict__ Qh,
    _Float16* __restrict__ Kh, _Float16* __restrict__ Vt) {
  const int idx = blockIdx.x * 256 + threadIdx.x;
  const int d  = idx & 63;
  const int t  = (idx >> 6) & (N_T - 1);
  const int bh = idx >> 17;          // 0..31
  const int b  = bh >> 3;
  const int h  = bh & 7;

  const size_t row = (size_t)(b * N_T + t) * N_3C;
  const int qc = h * N_HD + d;
  const int dp = (d < 32) ? d + 32 : d - 32;
  const float sgn = (d < 32) ? -1.0f : 1.0f;
  const float cv = cosT[t * N_HD + d];
  const float sv = sinT[t * N_HD + d];

  const float q  = (float)qkv[row + qc];
  const float q2 = (float)qkv[row + h * N_HD + dp];
  const float k  = (float)qkv[row + N_C + qc];
  const float k2 = (float)qkv[row + N_C + h * N_HD + dp];
  const float vv = (float)qkv[row + 2 * N_C + qc];

  Qh[((size_t)bh * N_T + t) * N_HD + d] = (_Float16)(q * cv + sgn * q2 * sv);
  Kh[((size_t)bh * N_T + t) * N_HD + d] = (_Float16)(k * cv + sgn * k2 * sv);
  Vt[((size_t)bh * N_HD + d) * N_T + t] = (_Float16)vv;
}

// ============================================================================
// Kernel 3: flash attention. One wave per (b,h, 16-query tile).
// Block = 128 threads (4 waves). grid = B*H*(T/16)/4 = 1024.
// Per 32-key chunk: 4 WMMA (QK^T, hd=64 = 2 K-steps x 2 key subtiles),
// online softmax (half-wave shuffles), P via LDS -> A-layout, 4 WMMA (PV).
// ============================================================================
__global__ __launch_bounds__(128) void attn_kernel(
    const _Float16* __restrict__ Qh, const _Float16* __restrict__ Kh,
    const _Float16* __restrict__ Vt, _Float16* __restrict__ O) {
  __shared__ float lds_p[4][16][32];

  const int lane  = threadIdx.x & 31;
  const int wslot = threadIdx.x >> 5;
  const int wid   = blockIdx.x * 4 + wslot;
  const int qt = wid & 127;          // query tile within (b,h)
  const int bh = wid >> 7;           // 0..31
  const int b  = bh >> 3;
  const int h  = bh & 7;
  const int qbase = qt * 16;
  const int grp = lane >> 4;
  const int l16 = lane & 15;
  const float scale = 0.125f;        // 1/sqrt(64)

  // Q A-fragments for hd=64: two 16x32 f16 fragments
  const _Float16* qrow = Qh + ((size_t)bh * N_T + qbase + l16) * N_HD;
  v16h aq[2];
#pragma unroll
  for (int di = 0; di < 2; ++di)
    aq[di] = load16h(qrow + di * 32 + grp * 8, qrow + di * 32 + 16 + grp * 8);

  v8f acc[4] = {};
  float mrow[8], lrow[8];
#pragma unroll
  for (int v = 0; v < 8; ++v) { mrow[v] = NEG_INF_F; lrow[v] = 0.0f; }

  const int nch = (qbase + 16 + 31) >> 5;   // 32-key chunks needed
  for (int ck = 0; ck < nch; ++ck) {
    const int kb = ck * 32;

    // ---- S = Q K^T (two 16x16 tiles: keys kb..kb+15 and kb+16..kb+31) ----
    v8f s[2] = {};
#pragma unroll
    for (int sub = 0; sub < 2; ++sub) {
      const int key = kb + sub * 16 + l16;  // B-matrix column = key index
      const _Float16* krow = Kh + ((size_t)bh * N_T + key) * N_HD;
#pragma unroll
      for (int di = 0; di < 2; ++di) {
        v16h bk = load16h(krow + di * 32 + grp * 8,
                          krow + di * 32 + 16 + grp * 8);
        s[sub] = wmma_f16(aq[di], bk, s[sub]);
      }
    }

    // ---- online softmax (rows v+8*grp live across the 16-lane group) ----
    float alpha[8];
#pragma unroll
    for (int v = 0; v < 8; ++v) {
      const int qrow_g = qbase + v + 8 * grp;
      float s0 = s[0][v] * scale;
      float s1 = s[1][v] * scale;
      if (kb + l16 > qrow_g)      s0 = NEG_INF_F;   // causal mask
      if (kb + 16 + l16 > qrow_g) s1 = NEG_INF_F;
      float rm = fmaxf(s0, s1);
#pragma unroll
      for (int msk = 1; msk < 16; msk <<= 1)
        rm = fmaxf(rm, __shfl_xor(rm, msk, 16));
      const float mn = fmaxf(mrow[v], rm);
      const float al = __expf(mrow[v] - mn);
      const float p0 = __expf(s0 - mn);
      const float p1 = __expf(s1 - mn);
      float rs = p0 + p1;
#pragma unroll
      for (int msk = 1; msk < 16; msk <<= 1)
        rs += __shfl_xor(rs, msk, 16);
      lrow[v] = lrow[v] * al + rs;
      mrow[v] = mn;
      alpha[v] = al;
      // stage P in LDS in C/D layout: [row][key-in-chunk]
      lds_p[wslot][v + 8 * grp][l16]      = p0;
      lds_p[wslot][v + 8 * grp][16 + l16] = p1;
    }
#pragma unroll
    for (int t = 0; t < 4; ++t)
#pragma unroll
      for (int v = 0; v < 8; ++v) acc[t][v] *= alpha[v];

    // LDS in-order per wave; wait for store data + block compiler reordering
    asm volatile("s_wait_dscnt 0x0" ::: "memory");

    // ---- re-shape P into 16x32 A-matrix layout ----
    union { v16h v; _Float16 e[16]; } pa;
#pragma unroll
    for (int i = 0; i < 8; ++i) {
      pa.e[i]     = (_Float16)lds_p[wslot][l16][grp * 8 + i];
      pa.e[8 + i] = (_Float16)lds_p[wslot][l16][16 + grp * 8 + i];
    }

    // ---- O += P @ V  (V transposed: column d contiguous over keys) ----
#pragma unroll
    for (int t = 0; t < 4; ++t) {
      const int dcol = t * 16 + l16;   // B-matrix column = hd index
      const _Float16* vcol =
          Vt + ((size_t)bh * N_HD + dcol) * N_T + kb;
      v16h bv = load16h(vcol + grp * 8, vcol + 16 + grp * 8);
      acc[t] = wmma_f16(pa.v, bv, acc[t]);
    }
  }

  // ---- normalize and store to [B,T,C] f16 (heads re-interleaved) ----
#pragma unroll
  for (int v = 0; v < 8; ++v) {
    const float inv = 1.0f / lrow[v];
    const int q = qbase + v + 8 * grp;
#pragma unroll
    for (int t = 0; t < 4; ++t) {
      const int c = h * N_HD + t * 16 + l16;
      O[((size_t)b * N_T + q) * N_C + c] = (_Float16)(acc[t][v] * inv);
    }
  }
}

// ============================================================================
// Kernel 4: out = attn @ W_out^T  (M=8192, N=512, K=512), all-f16, f32 out
// Block = 256 threads (8 waves), wave does 32x64; grid = M/32 = 256.
// ============================================================================
__global__ __launch_bounds__(256) void out_gemm_kernel(
    const _Float16* __restrict__ A, const _Float16* __restrict__ W,
    float* __restrict__ Y) {
  const int lane = threadIdx.x & 31;
  const int wave = threadIdx.x >> 5;
  const int grp  = lane >> 4;
  const int l16  = lane & 15;
  const int row0 = blockIdx.x * 32;
  const int col0 = wave * 64;

  v8f acc0[4] = {}, acc1[4] = {};
  const _Float16* arow0 = A + (size_t)(row0 + l16) * N_C;
  const _Float16* arow1 = A + (size_t)(row0 + 16 + l16) * N_C;

  for (int k0 = 0; k0 < N_C; k0 += 32) {
    __builtin_prefetch(arow0 + k0 + 128, 0, 1);
    v16h a0 = load16h(arow0 + k0 + grp * 8, arow0 + k0 + 16 + grp * 8);
    v16h a1 = load16h(arow1 + k0 + grp * 8, arow1 + k0 + 16 + grp * 8);
#pragma unroll
    for (int n = 0; n < 4; ++n) {
      const _Float16* wrow =
          W + (size_t)(col0 + n * 16 + l16) * N_C + k0;
      v16h b = load16h(wrow + grp * 8, wrow + 16 + grp * 8);
      acc0[n] = wmma_f16(a0, b, acc0[n]);
      acc1[n] = wmma_f16(a1, b, acc1[n]);
    }
  }
#pragma unroll
  for (int n = 0; n < 4; ++n) {
#pragma unroll
    for (int v = 0; v < 8; ++v) {
      const int r = row0 + v + grp * 8;
      const int c = col0 + n * 16 + l16;
      Y[(size_t)r * N_C + c] = acc0[n][v];
      Y[(size_t)(r + 16) * N_C + c] = acc1[n][v];
    }
  }
}

// ============================================================================
extern "C" void kernel_launch(void* const* d_in, const int* in_sizes, int n_in,
                              void* d_out, int out_size, void* d_ws,
                              size_t ws_size, hipStream_t stream) {
  const float* x     = (const float*)d_in[0];   // (4,2048,512)
  const float* cosT  = (const float*)d_in[1];   // (2048,64)
  const float* sinT  = (const float*)d_in[2];   // (2048,64)
  const float* W_qkv = (const float*)d_in[3];   // (1536,512)
  const float* W_out = (const float*)d_in[4];   // (512,512)
  float* out = (float*)d_out;                   // (4,2048,512) f32

  // workspace layout (all f16):
  // qkv | Qh | Kh | Vt | attn | xh | Wqh | Woh   (~67 MB)
  const size_t n_x   = (size_t)N_B * N_T * N_C;        // 4194304
  const size_t n_qkv = (size_t)N_B * N_T * N_3C;       // 12582912
  const size_t n_hd  = (size_t)N_B * N_H * N_T * N_HD; // 4194304
  const size_t n_wq  = (size_t)N_3C * N_C;             // 786432
  const size_t n_wo  = (size_t)N_C * N_C;              // 262144

  _Float16* qkvh = (_Float16*)d_ws;
  _Float16* Qh   = qkvh + n_qkv;
  _Float16* Kh   = Qh + n_hd;
  _Float16* Vt   = Kh + n_hd;
  _Float16* attn = Vt + n_hd;
  _Float16* xh   = attn + n_x;
  _Float16* Wqh  = xh + n_x;
  _Float16* Woh  = Wqh + n_wq;

  // 0) one-shot f32 -> f16 conversions
  cvt_f16_kernel<<<(int)(n_x / 4 + 255) / 256, 256, 0, stream>>>(
      x, xh, (int)(n_x / 4));
  cvt_f16_kernel<<<(int)(n_wq / 4 + 255) / 256, 256, 0, stream>>>(
      W_qkv, Wqh, (int)(n_wq / 4));
  cvt_f16_kernel<<<(int)(n_wo / 4 + 255) / 256, 256, 0, stream>>>(
      W_out, Woh, (int)(n_wo / 4));

  // 1) QKV projection: (8192x512) x (1536x512)^T, all-f16 operands
  qkv_gemm_kernel<<<dim3(256, 3), 256, 0, stream>>>(xh, Wqh, qkvh);

  // 2) RoPE + repack (Q,K rotated; V transposed per head)
  rope_repack_kernel<<<(N_B * N_H * N_T * N_HD) / 256, 256, 0, stream>>>(
      qkvh, cosT, sinT, Qh, Kh, Vt);

  // 3) causal flash attention, one wave per 16-query tile
  attn_kernel<<<1024, 128, 0, stream>>>(Qh, Kh, Vt, attn);

  // 4) output projection: (8192x512) x (512x512)^T -> f32
  out_gemm_kernel<<<256, 256, 0, stream>>>(attn, Woh, out);
}
